// AttnGRU_68977174774224
// MI455X (gfx1250) — matrix operand order repaired and checked by reading, not compiled, not run on hardware
//
#include <hip/hip_runtime.h>

// ---------------------------------------------------------------------------
// AttnGRU on MI455X (gfx1250).
//   r   = sigmoid( x_t@Wr^T + Wr_b + h@Ur^T + Ur_b )
//   n   = tanh   ( x_t@W^T  + W_b  + r*(h@U^T + U_b) )
//   h'  = (1-g)*n + g*h
// bf16 WMMA (v_wmma_f32_16x16x32_bf16) with f32 accumulate. x and weights
// pre-converted to bf16 once; h kept in f32 (epilogue) + bf16 (GEMM feed).
// A tiles double-buffered in LDS and filled with CDNA5 async global->LDS
// copies (ASYNCcnt) so staging overlaps the WMMA stream.
// ---------------------------------------------------------------------------

typedef __attribute__((ext_vector_type(16))) __bf16 v16bf;
typedef __attribute__((ext_vector_type(8)))  float  v8f;
typedef __attribute__((ext_vector_type(4)))  int    v4i;

#define GLOBAL_AS __attribute__((address_space(1)))
#define LDS_AS    __attribute__((address_space(3)))

#define Hdim   1024
#define Bsz    256
#define Tsteps 128
#define KC     128           // K chunk staged in LDS (8 chunks, double-buffered)
#define NCHUNK (Hdim / KC)
#define LDA    (KC + 8)      // padded LDS row stride in halves (16B aligned)

__device__ __forceinline__ unsigned short f2bf(float f) {
  unsigned int u = __float_as_uint(f);
  u += 0x7fffu + ((u >> 16) & 1u);   // round-to-nearest-even
  return (unsigned short)(u >> 16);
}

union FragBF { uint4 q[2]; v16bf v; };

// 16B/lane async global->LDS copy (ASYNCcnt); sync fallback if unavailable.
__device__ __forceinline__ void copy16(const unsigned short* g, unsigned short* l) {
#if __has_builtin(__builtin_amdgcn_global_load_async_to_lds_b128)
  __builtin_amdgcn_global_load_async_to_lds_b128(
      (GLOBAL_AS v4i*)(uintptr_t)g,   // AS1 int4* (inttoptr: same repr)
      (LDS_AS v4i*)l,                 // AS3 int4* (addrspacecast)
      0, 0);
#else
  *(uint4*)l = *(const uint4*)g;
#endif
}

__device__ __forceinline__ void wait_async() {
#if __has_builtin(__builtin_amdgcn_s_wait_asynccnt)
  __builtin_amdgcn_s_wait_asynccnt(0);
#else
  asm volatile("s_wait_asynccnt 0" ::: "memory");
#endif
}

// -------- generic f32 -> bf16 conversion (x, h0) ---------------------------
__global__ void cvt_f32_bf16_kernel(const float* __restrict__ src,
                                    unsigned short* __restrict__ dst, int n4) {
  for (int i = blockIdx.x * blockDim.x + threadIdx.x; i < n4;
       i += gridDim.x * blockDim.x) {
    const float4 v = ((const float4*)src)[i];
    uint2 p;
    p.x = (unsigned)f2bf(v.x) | ((unsigned)f2bf(v.y) << 16);
    p.y = (unsigned)f2bf(v.z) | ((unsigned)f2bf(v.w) << 16);
    ((uint2*)dst)[i] = p;
  }
}

// -------- one-time weight conversion: 4x [H,H] f32 -> bf16 ------------------
__global__ void convert_weights_kernel(const float* __restrict__ Wr,
                                       const float* __restrict__ Ur,
                                       const float* __restrict__ W,
                                       const float* __restrict__ U,
                                       unsigned short* __restrict__ dst) {
  const int total = Hdim * Hdim;
  for (int idx = blockIdx.x * blockDim.x + threadIdx.x; idx < total;
       idx += gridDim.x * blockDim.x) {
    dst[0 * total + idx] = f2bf(Wr[idx]);
    dst[1 * total + idx] = f2bf(Ur[idx]);
    dst[2 * total + idx] = f2bf(W[idx]);
    dst[3 * total + idx] = f2bf(U[idx]);
  }
}

// -------- one recurrence step ----------------------------------------------
// 128 threads = 4 wave32s. Block tile 32(M) x 64(N); wave = one 16-wide N
// tile; each wave accumulates TWO 16-row M subtiles so every B fragment
// feeds 2 WMMAs. LDS A-tiles double-buffered: async copies for chunk kc+1
// run underneath the WMMAs of chunk kc.
__global__ __launch_bounds__(128)
void gru_step_kernel(const unsigned short* __restrict__ xbf,   // [B,T,H] bf16
                     const float* __restrict__ h_in,           // [B,H] f32
                     const unsigned short* __restrict__ hbf_in,// [B,H] bf16
                     float* __restrict__ h_out,                // [B,H] f32
                     unsigned short* __restrict__ hbf_out,     // [B,H] bf16
                     const unsigned short* __restrict__ wb,    // bf16 Wr,Ur,W,U
                     const float* __restrict__ Wr_b,
                     const float* __restrict__ Ur_b,
                     const float* __restrict__ W_b,
                     const float* __restrict__ U_b,
                     const float* __restrict__ gt,             // [B]
                     int t) {
  __shared__ unsigned short Ax[2][32 * LDA];  // bf16 x_t tiles (32 x KC) x2
  __shared__ unsigned short Ah[2][32 * LDA];  // bf16 h   tiles (32 x KC) x2

  const int tid  = threadIdx.x;
  const int lane = tid & 31;
  const int wn   = tid >> 5;                  // 4 N tiles (block N = 64)
  const int mBlock = blockIdx.x * 32;         // batch rows b
  const int nBlock = blockIdx.y * 64;         // output cols o

  const unsigned short* wWr = wb + 0ull * Hdim * Hdim;
  const unsigned short* wUr = wb + 1ull * Hdim * Hdim;
  const unsigned short* wW  = wb + 2ull * Hdim * Hdim;
  const unsigned short* wU  = wb + 3ull * Hdim * Hdim;

  v8f axr[2] = {{}, {}}, ahr[2] = {{}, {}}, axn[2] = {{}, {}}, ahn[2] = {{}, {}};

  const int halfsel = (lane < 16) ? 0 : 1;
  const int row0    = lane & 15;                     // A subtile-0 row
  const int nCol    = nBlock + wn * 16 + (lane & 15);// B row / D column

  // issue async copies of chunk kc into LDS buffer `buf`
  auto stage = [&](int kc, int buf) {
    const int k0 = kc * KC;
    for (int e = tid * 8; e < 32 * KC; e += 128 * 8) {
      const int r = e / KC;
      const int c = e % KC;
      copy16(&xbf[((size_t)(mBlock + r) * Tsteps + t) * Hdim + k0 + c],
             &Ax[buf][r * LDA + c]);
      copy16(&hbf_in[(size_t)(mBlock + r) * Hdim + k0 + c],
             &Ah[buf][r * LDA + c]);
    }
  };

  stage(0, 0);   // prologue: fill buffer 0

  #pragma unroll 2
  for (int kc = 0; kc < NCHUNK; ++kc) {
    const int buf = kc & 1;
    const int k0  = kc * KC;
    wait_async();        // drain copies issued one iteration ago
    __syncthreads();     // make them visible to all waves / protect WAR
    if (kc + 1 < NCHUNK) stage(kc + 1, buf ^ 1);  // prefetch next chunk

    // ---- 4 WMMA K-steps of 32 over this chunk ----
    #pragma unroll
    for (int ks = 0; ks < KC; ks += 32) {
      const int ca = ks + halfsel * 8;   // A: lane<16 K{0..7,16..23}, else +8
      FragBF ax0, ax1, ah0, ah1;
      ax0.q[0] = *(const uint4*)&Ax[buf][row0 * LDA + ca];
      ax0.q[1] = *(const uint4*)&Ax[buf][row0 * LDA + ca + 16];
      ax1.q[0] = *(const uint4*)&Ax[buf][(row0 + 16) * LDA + ca];
      ax1.q[1] = *(const uint4*)&Ax[buf][(row0 + 16) * LDA + ca + 16];
      ah0.q[0] = *(const uint4*)&Ah[buf][row0 * LDA + ca];
      ah0.q[1] = *(const uint4*)&Ah[buf][row0 * LDA + ca + 16];
      ah1.q[0] = *(const uint4*)&Ah[buf][(row0 + 16) * LDA + ca];
      ah1.q[1] = *(const uint4*)&Ah[buf][(row0 + 16) * LDA + ca + 16];

      // B fragments: weight rows are K-contiguous -> two b128 per matrix
      const size_t kb = (size_t)nCol * Hdim + (k0 + ks) + halfsel * 8;
      FragBF bwr, bur, bw, bu;
      bwr.q[0] = *(const uint4*)&wWr[kb]; bwr.q[1] = *(const uint4*)&wWr[kb + 16];
      bur.q[0] = *(const uint4*)&wUr[kb]; bur.q[1] = *(const uint4*)&wUr[kb + 16];
      bw.q[0]  = *(const uint4*)&wW[kb];  bw.q[1]  = *(const uint4*)&wW[kb + 16];
      bu.q[0]  = *(const uint4*)&wU[kb];  bu.q[1]  = *(const uint4*)&wU[kb + 16];

      axr[0] = __builtin_amdgcn_wmma_f32_16x16x32_bf16(false, ax0.v, false, bwr.v, (short)0, axr[0], false, false);
      axr[1] = __builtin_amdgcn_wmma_f32_16x16x32_bf16(false, ax1.v, false, bwr.v, (short)0, axr[1], false, false);
      ahr[0] = __builtin_amdgcn_wmma_f32_16x16x32_bf16(false, ah0.v, false, bur.v, (short)0, ahr[0], false, false);
      ahr[1] = __builtin_amdgcn_wmma_f32_16x16x32_bf16(false, ah1.v, false, bur.v, (short)0, ahr[1], false, false);
      axn[0] = __builtin_amdgcn_wmma_f32_16x16x32_bf16(false, ax0.v, false, bw.v,  (short)0, axn[0], false, false);
      axn[1] = __builtin_amdgcn_wmma_f32_16x16x32_bf16(false, ax1.v, false, bw.v,  (short)0, axn[1], false, false);
      ahn[0] = __builtin_amdgcn_wmma_f32_16x16x32_bf16(false, ah0.v, false, bu.v,  (short)0, ahn[0], false, false);
      ahn[1] = __builtin_amdgcn_wmma_f32_16x16x32_bf16(false, ah1.v, false, bu.v,  (short)0, ahn[1], false, false);
    }
  }

  // ---- f32 gate epilogue (writes f32 + bf16 copies of h') ----
  const float brW = Wr_b[nCol];
  const float brU = Ur_b[nCol];
  const float bnW = W_b[nCol];
  const float bnU = U_b[nCol];
  #pragma unroll
  for (int s = 0; s < 2; ++s) {
    #pragma unroll
    for (int i = 0; i < 8; ++i) {
      const int m = s * 16 + i + halfsel * 8;  // C/D layout: vgpr i -> M=i(+8)
      const int b = mBlock + m;
      const float g     = gt[b];
      const float hprev = h_in[(size_t)b * Hdim + nCol];
      const float pre_r = axr[s][i] + ahr[s][i] + brW + brU;
      const float r     = 1.0f / (1.0f + __expf(-pre_r));
      const float pre_n = axn[s][i] + bnW + r * (ahn[s][i] + bnU);
      const float nv    = tanhf(pre_n);
      const float hnew  = (1.0f - g) * nv + g * hprev;
      h_out[(size_t)b * Hdim + nCol]   = hnew;
      hbf_out[(size_t)b * Hdim + nCol] = f2bf(hnew);
    }
  }
}

extern "C" void kernel_launch(void* const* d_in, const int* in_sizes, int n_in,
                              void* d_out, int out_size, void* d_ws, size_t ws_size,
                              hipStream_t stream) {
  const float* x    = (const float*)d_in[0];
  const float* h0   = (const float*)d_in[1];
  const float* gt   = (const float*)d_in[2];
  const float* Wr_w = (const float*)d_in[3];
  const float* Wr_b = (const float*)d_in[4];
  const float* Ur_w = (const float*)d_in[5];
  const float* Ur_b = (const float*)d_in[6];
  // d_in[7..10] = Wz_w, Wz_b, Uz_w, Uz_b  (dead in the reference)
  const float* W_w  = (const float*)d_in[11];
  const float* W_b  = (const float*)d_in[12];
  const float* U_w  = (const float*)d_in[13];
  const float* U_b  = (const float*)d_in[14];

  float* out = (float*)d_out;
  char* ws = (char*)d_ws;
  // workspace layout
  unsigned short* xbf  = (unsigned short*)(ws);                       // 64 MB
  unsigned short* wbf  = (unsigned short*)(ws + 67108864ull);         //  8 MB
  float*          hbuf = (float*)(ws + 67108864ull + 8388608ull);     //  1 MB
  unsigned short* hbf0 = (unsigned short*)(ws + 75497472ull + 1048576ull);
  unsigned short* hbf1 = hbf0 + (size_t)Bsz * Hdim;                   // 2x 0.5MB

  convert_weights_kernel<<<512, 256, 0, stream>>>(Wr_w, Ur_w, W_w, U_w, wbf);
  cvt_f32_bf16_kernel<<<2048, 256, 0, stream>>>(x, xbf, Bsz * Tsteps * Hdim / 4);
  cvt_f32_bf16_kernel<<<64, 256, 0, stream>>>(h0, hbf0, Bsz * Hdim / 4);

  dim3 grid(Bsz / 32, Hdim / 64);   // 8 x 16 workgroups of 4 wave32s
  for (int t = 0; t < Tsteps; ++t) {
    // f32 ping-pong arranged so step T-1 (odd) writes d_out
    const float* hin  = (t == 0) ? h0 : ((t & 1) ? (const float*)hbuf
                                                 : (const float*)out);
    float* hout = (t & 1) ? out : hbuf;
    const unsigned short* hbin = (t & 1) ? hbf1 : hbf0;
    unsigned short* hbout      = (t & 1) ? hbf0 : hbf1;
    gru_step_kernel<<<grid, 128, 0, stream>>>(xbf, hin, hbin, hout, hbout, wbf,
                                              Wr_b, Ur_b, W_b, U_b, gt, t);
  }
}
